// ConvGPTAttention_83167746720363
// MI455X (gfx1250) — matrix operand order repaired and checked by reading, not compiled, not run on hardware
//
#include <hip/hip_runtime.h>
#include <hip/hip_bf16.h>
#include <stdint.h>

// ---------------------------------------------------------------------------
// CDNA5 (gfx1250) attention block: QKV GEMM -> RMSNorm+RoPE -> flash attn ->
// O GEMM. Matmuls via v_wmma_f32_16x16x32_bf16 (fp32 accum). Projection GEMMs
// use LDS-staged 128x128 tiles with double buffering; tile staging goes
// through GLOBAL_LOAD_ASYNC_TO_LDS_B128 (ASYNCcnt-tracked DMA).
// ---------------------------------------------------------------------------

typedef __attribute__((ext_vector_type(16))) __bf16 v16bf;
typedef __attribute__((ext_vector_type(8)))  float  v8f;

union AFrag { v16bf v; uint32_t u[8]; };

__device__ __forceinline__ uint16_t f2bf(float f) {
  uint32_t u = __float_as_uint(f);
  u += 0x7FFFu + ((u >> 16) & 1u);   // round-to-nearest-even
  return (uint16_t)(u >> 16);
}

// A-matrix 16x32 bf16 per-lane K-pair base (ISA 7.12.2): lanes 0-15 hold
// K={0..7,16..23}, lanes 16-31 hold K={8..15,24..31}, 2 K per VGPR.
__device__ __forceinline__ int a_kpair(int t, int khalf) {
  return 2 * t + (t >= 4 ? 8 : 0) + 8 * khalf;
}
// B-matrix 32x16: same K order as A: lanes 0-15 hold K={0..7,16..23}, lanes 16-31 K={8..15,24..31}.
__device__ __forceinline__ int b_kpair(int t, int khalf) {
  return 2 * t + (t >= 4 ? 8 : 0) + 8 * khalf;
}

// --------------------- async global->LDS staging path ----------------------

#if defined(__has_builtin)
#if __has_builtin(__builtin_amdgcn_global_load_async_to_lds_b128)
#define HAVE_ASYNC_LDS 1
#endif
#endif
#ifndef HAVE_ASYNC_LDS
#define HAVE_ASYNC_LDS 0
#endif

#if HAVE_ASYNC_LDS
// Builtin signature (from compiler diagnostic): param0 is
// 'int __vector_size__(16) __device__ *' i.e. GCC-vector int4 in AS1 (global);
// param1 is the LDS-side int4 pointer (AS3); then imm offset, imm cpol.
typedef int v4i_vs __attribute__((vector_size(16)));
typedef __attribute__((address_space(1))) v4i_vs* as1_v4i;
typedef __attribute__((address_space(3))) v4i_vs* as3_v4i;
__device__ __forceinline__ void cp_async16(const void* g, void* l) {
  __builtin_amdgcn_global_load_async_to_lds_b128(
      (as1_v4i)(uintptr_t)g, (as3_v4i)(uint32_t)(uintptr_t)l, 0, 0);
}
#if __has_builtin(__builtin_amdgcn_s_wait_asynccnt)
#define WAIT_ASYNC() __builtin_amdgcn_s_wait_asynccnt(0)
#else
#define WAIT_ASYNC() asm volatile("s_wait_asynccnt 0x0" ::: "memory")
#endif
#else
__device__ __forceinline__ void cp_async16(const void* g, void* l) {
  *(uint4*)l = *(const uint4*)g;     // vmem load + ds_store fallback
}
#define WAIT_ASYNC() ((void)0)
#endif

// ------------------------- conversion prepasses ----------------------------

__global__ void cvt_bf16(const float* __restrict__ src, uint16_t* __restrict__ dst, int n) {
  int i = blockIdx.x * 256 + threadIdx.x;
  if (i < n) dst[i] = f2bf(src[i]);
}

// src [R][C] fp32 -> dst [C][R] bf16. grid (C/256, R), block 256.
__global__ void cvt_transpose_bf16(const float* __restrict__ src, uint16_t* __restrict__ dst,
                                   int R, int C) {
  int c = blockIdx.x * 256 + threadIdx.x;
  int r = blockIdx.y;
  if (c < C) dst[(size_t)c * R + r] = f2bf(src[(size_t)r * C + c]);
}

// V slice of qkv [T][4096] (cols 3072..4095) -> vT [8][128][T] bf16
__global__ void cvt_v(const float* __restrict__ qkv, uint16_t* __restrict__ vT, int T, int total) {
  int i = blockIdx.x * 256 + threadIdx.x;
  if (i >= total) return;
  int t = i >> 10;          // / 1024
  int r = i & 1023;         // h*128 + d
  vT[(size_t)r * T + t] = f2bf(qkv[(size_t)t * 4096 + 3072 + r]);
}

// ------------------------------ bf16 GEMM ----------------------------------
// C[M][N] fp32 = A[M][K] bf16 * Bt[N][K]^T bf16.
// Block = 128 threads (4 waves), block tile 128x128, wave tile 64x64
// (4x4 WMMA frags). K step 32, LDS double-buffered, async staging.
__global__ void __launch_bounds__(128)
gemm_bf16_nt(const uint16_t* __restrict__ A, const uint16_t* __restrict__ Bt,
             float* __restrict__ C, int M, int N, int K) {
  __shared__ __align__(16) uint16_t As[2][128][32];
  __shared__ __align__(16) uint16_t Bs[2][128][32];

  const int tid   = threadIdx.x;
  const int wave  = tid >> 5;
  const int lane  = tid & 31;
  const int khalf = lane >> 4;
  const int mn    = lane & 15;
  const int M0 = blockIdx.y * 128;
  const int N0 = blockIdx.x * 128;
  const int wr = (wave >> 1) * 64;     // wave row offset in block tile
  const int wc = (wave & 1) * 64;      // wave col offset in block tile

  v8f acc[4][4];
#pragma unroll
  for (int i = 0; i < 4; ++i)
#pragma unroll
    for (int j = 0; j < 4; ++j) acc[i][j] = (v8f){0, 0, 0, 0, 0, 0, 0, 0};

  // Stage one 128x32 A tile + 128x32 B tile (8KB each): 4x16B chunks/thread.
  auto stage = [&](int buf, int kt) {
#pragma unroll
    for (int p = 0; p < 4; ++p) {
      const int c   = p * 128 + tid;
      const int row = c >> 2;
      const int kc  = (c & 3) * 8;     // element offset (16B chunks)
      cp_async16(A  + (size_t)(M0 + row) * K + kt + kc, &As[buf][row][kc]);
      cp_async16(Bt + (size_t)(N0 + row) * K + kt + kc, &Bs[buf][row][kc]);
    }
  };

  int buf = 0;
  stage(0, 0);
  WAIT_ASYNC();
  __syncthreads();

  for (int kt = 0; kt < K; kt += 32) {
    if (kt + 32 < K) stage(buf ^ 1, kt + 32);   // overlap DMA with WMMA

    AFrag af[4], bfr[4];
#pragma unroll
    for (int i = 0; i < 4; ++i) {
      const uint16_t* ar = &As[buf][wr + 16 * i + mn][0];
#pragma unroll
      for (int t = 0; t < 8; ++t) af[i].u[t] = *(const uint32_t*)(ar + a_kpair(t, khalf));
      const uint16_t* br = &Bs[buf][wc + 16 * i + mn][0];
#pragma unroll
      for (int t = 0; t < 8; ++t) bfr[i].u[t] = *(const uint32_t*)(br + b_kpair(t, khalf));
    }
#pragma unroll
    for (int i = 0; i < 4; ++i)
#pragma unroll
      for (int j = 0; j < 4; ++j)
        acc[i][j] = __builtin_amdgcn_wmma_f32_16x16x32_bf16(
            false, af[i].v, false, bfr[j].v, (short)0, acc[i][j], false, false);

    WAIT_ASYNC();
    __syncthreads();
    buf ^= 1;
  }

#pragma unroll
  for (int i = 0; i < 4; ++i)
#pragma unroll
    for (int j = 0; j < 4; ++j)
#pragma unroll
      for (int r = 0; r < 8; ++r) {
        const int mA = M0 + wr + 16 * i + 8 * khalf + r;
        C[(size_t)mA * N + N0 + wc + 16 * j + mn] = acc[i][j][r];
      }
}

// ----------------------- per-head RMSNorm + RoPE ---------------------------
// grid (T, 24): j<16 -> q head j, else k head j-16. 128 threads = one head row.
__global__ void __launch_bounds__(128)
normrope(const float* __restrict__ qkv, const int* __restrict__ pos,
         const float* __restrict__ qw, const float* __restrict__ kw,
         uint16_t* __restrict__ qb, uint16_t* __restrict__ kb, int T) {
  __shared__ float xs[128];
  __shared__ float red[4];
  const int t = blockIdx.x, j = blockIdx.y, i = threadIdx.x;
  const bool isq = (j < 16);
  const int h = isq ? j : j - 16;
  const float* row = qkv + (size_t)t * 4096 + (isq ? h * 128 : 2048 + h * 128);

  float x  = row[i];
  float ss = x * x;
#pragma unroll
  for (int off = 16; off > 0; off >>= 1) ss += __shfl_xor(ss, off, 32);
  if ((i & 31) == 0) red[i >> 5] = ss;
  __syncthreads();
  ss = red[0] + red[1] + red[2] + red[3];

  const float rms = rsqrtf(ss * (1.0f / 128.0f) + 1e-6f);
  const float xn  = x * rms * (isq ? qw[i] : kw[i]);
  xs[i] = xn;
  __syncthreads();

  const int   fi = i & 63;
  const float p  = (float)pos[t];
  const float inv_freq = __powf(10000.0f, -(float)fi * (1.0f / 64.0f));
  const float fr = p * inv_freq;
  const float c  = __cosf(fr), s = __sinf(fr);
  const float out = (i < 64) ? (xn * c - xs[i + 64] * s)
                             : (xn * c + xs[i - 64] * s);
  const uint16_t ob = f2bf(out);
  if (isq) qb[((size_t)t * 16 + h) * 128 + i] = ob;          // [T][16][128]
  else     kb[((size_t)h * T + t) * 128 + i] = ob;           // [8][T][128]
}

// --------------------------- flash attention -------------------------------
// Block = 4 waves; wave w handles 16 query rows q0 = blockIdx.x*64 + w*16 of
// head blockIdx.y. Online softmax, causal, key tiles of 32. Wave-private trip
// counts (no block barriers); P relayout via per-wave LDS + s_wait_dscnt.
__global__ void __launch_bounds__(128)
flash_attn(const uint16_t* __restrict__ qb, const uint16_t* __restrict__ kbuf,
           const uint16_t* __restrict__ vT, uint16_t* __restrict__ ob, int T) {
  __shared__ uint16_t plds[4][16][32];
  const int wave  = threadIdx.x >> 5;
  const int lane  = threadIdx.x & 31;
  const int khalf = lane >> 4;
  const int mn    = lane & 15;
  const int h   = blockIdx.y;
  const int q0  = blockIdx.x * 64 + wave * 16;
  const int kvh = h >> 1;                      // GQA: 16 q heads -> 8 kv heads

  // Q fragments: 16 rows x 128 hd as 4 A-frags (held in registers)
  AFrag qf[4];
  const uint16_t* qrow = qb + ((size_t)(q0 + mn) * 16 + h) * 128;
#pragma unroll
  for (int f = 0; f < 4; ++f)
#pragma unroll
    for (int t = 0; t < 8; ++t)
      qf[f].u[t] = *(const uint32_t*)(qrow + 32 * f + a_kpair(t, khalf));

  v8f o[8];
  float m_i[8], l_i[8];
#pragma unroll
  for (int f = 0; f < 8; ++f) o[f] = (v8f){0, 0, 0, 0, 0, 0, 0, 0};
#pragma unroll
  for (int r = 0; r < 8; ++r) { m_i[r] = -1e30f; l_i[r] = 0.0f; }

  // hoisted base pointers (K rows for this lane's key slot; V rows per hd chunk)
  const uint16_t* kbase0 = kbuf + ((size_t)kvh * T + mn) * 128;        // key kb0+mn
  const uint16_t* kbase1 = kbuf + ((size_t)kvh * T + 16 + mn) * 128;   // key kb0+16+mn
  const uint16_t* vbase[8];
#pragma unroll
  for (int f = 0; f < 8; ++f)
    vbase[f] = vT + ((size_t)kvh * 128 + f * 16 + mn) * T;

  const float sc = 0.08838834764831845f * 1.4426950408889634f; // 1/sqrt(128) * log2(e)
  const int nkt = (q0 + 15) / 32 + 1;          // causal tile count for this wave

  for (int kt = 0; kt < nkt; ++kt) {
    const int kb0 = kt * 32;
    v8f s0 = {}, s1 = {};
    const uint16_t* kr0b = kbase0 + (size_t)kb0 * 128;
    const uint16_t* kr1b = kbase1 + (size_t)kb0 * 128;
    // S = Q K^T over hd=128 (4 K-steps), 32 keys as two N-halves
#pragma unroll
    for (int f = 0; f < 4; ++f) {
      AFrag b0, b1;
#pragma unroll
      for (int t = 0; t < 8; ++t) {
        const int kk = 32 * f + b_kpair(t, khalf);
        b0.u[t] = *(const uint32_t*)(kr0b + kk);
        b1.u[t] = *(const uint32_t*)(kr1b + kk);
      }
      s0 = __builtin_amdgcn_wmma_f32_16x16x32_bf16(false, qf[f].v, false, b0.v, (short)0, s0, false, false);
      s1 = __builtin_amdgcn_wmma_f32_16x16x32_bf16(false, qf[f].v, false, b1.v, (short)0, s1, false, false);
    }

    // Online softmax in exp2 domain; row M = 8*khalf + r lives on 16 lanes.
    float al[8];
#pragma unroll
    for (int r = 0; r < 8; ++r) {
      const int qr = q0 + 8 * khalf + r;
      float z0 = (kb0 + mn      > qr) ? -1e30f : s0[r] * sc;
      float z1 = (kb0 + 16 + mn > qr) ? -1e30f : s1[r] * sc;
      float mx = fmaxf(z0, z1);
#pragma unroll
      for (int off = 1; off < 16; off <<= 1) mx = fmaxf(mx, __shfl_xor(mx, off, 32));
      const float mnew = fmaxf(m_i[r], mx);
      al[r] = exp2f(m_i[r] - mnew);
      const float p0 = exp2f(z0 - mnew);
      const float p1 = exp2f(z1 - mnew);
      float ls = p0 + p1;
#pragma unroll
      for (int off = 1; off < 16; off <<= 1) ls += __shfl_xor(ls, off, 32);
      l_i[r] = l_i[r] * al[r] + ls;
      m_i[r] = mnew;
      const int M = 8 * khalf + r;
      plds[wave][M][mn]      = f2bf(p0);       // C-layout -> LDS (A-layout reload)
      plds[wave][M][16 + mn] = f2bf(p1);
    }
    asm volatile("s_wait_dscnt 0x0" ::: "memory");   // wave-private LDS ordering

    AFrag pa;
#pragma unroll
    for (int t = 0; t < 8; ++t)
      pa.u[t] = *(const uint32_t*)&plds[wave][mn][a_kpair(t, khalf)];

    // O = diag(alpha) * O + P * V   (V pre-transposed: vT[kvh][hd][T])
#pragma unroll
    for (int f = 0; f < 8; ++f) {
#pragma unroll
      for (int r = 0; r < 8; ++r) o[f][r] *= al[r];
      AFrag bv;
      const uint16_t* vrow = vbase[f] + kb0;
#pragma unroll
      for (int t = 0; t < 8; ++t)
        bv.u[t] = *(const uint32_t*)(vrow + b_kpair(t, khalf));
      o[f] = __builtin_amdgcn_wmma_f32_16x16x32_bf16(false, pa.v, false, bv.v, (short)0, o[f], false, false);
    }
  }

  // epilogue: O / l, write bf16 [T][2048] for the output GEMM
#pragma unroll
  for (int f = 0; f < 8; ++f) {
#pragma unroll
    for (int r = 0; r < 8; ++r) {
      const int trow = q0 + 8 * khalf + r;
      const float val = o[f][r] / l_i[r];
      ob[(size_t)trow * 2048 + h * 128 + f * 16 + mn] = f2bf(val);
    }
  }
}

// ------------------------------- launcher ----------------------------------

extern "C" void kernel_launch(void* const* d_in, const int* in_sizes, int n_in,
                              void* d_out, int out_size, void* d_ws, size_t ws_size,
                              hipStream_t stream) {
  const int*   positions = (const int*)d_in[0];
  const float* hidden    = (const float*)d_in[1];
  const float* Wqkv      = (const float*)d_in[2];
  const float* Wo        = (const float*)d_in[3];
  const float* qw        = (const float*)d_in[4];
  const float* kw        = (const float*)d_in[5];
  float* out = (float*)d_out;

  const int T   = in_sizes[0];          // 2048
  const int H   = in_sizes[1] / T;      // 2048
  const int QKV = in_sizes[2] / H;      // 4096

  uint8_t* ws = (uint8_t*)d_ws;
  size_t off = 0;
  auto walloc = [&](size_t bytes) -> void* {
    void* p = ws + off;
    off += (bytes + 255) & ~(size_t)255;
    return p;
  };
  uint16_t* hb   = (uint16_t*)walloc((size_t)T * H * 2);          // hidden bf16
  uint16_t* WqT  = (uint16_t*)walloc((size_t)QKV * H * 2);        // Wqkv^T bf16
  float*    qkv  = (float*)   walloc((size_t)T * QKV * 4);        // QKV fp32
  uint16_t* qb   = (uint16_t*)walloc((size_t)T * 16 * 128 * 2);   // q [T][16][128]
  uint16_t* kbuf = (uint16_t*)walloc((size_t)8 * T * 128 * 2);    // k [8][T][128]
  uint16_t* vT   = (uint16_t*)walloc((size_t)8 * 128 * T * 2);    // v [8][128][T]
  uint16_t* ab   = (uint16_t*)walloc((size_t)T * 2048 * 2);       // attn out bf16
  uint16_t* WoT  = (uint16_t*)walloc((size_t)H * 2048 * 2);       // Wo^T bf16

  cvt_bf16<<<(T * H + 255) / 256, 256, 0, stream>>>(hidden, hb, T * H);
  {
    dim3 gt((QKV + 255) / 256, H);
    cvt_transpose_bf16<<<gt, 256, 0, stream>>>(Wqkv, WqT, H, QKV);
  }
  {
    dim3 gt((H + 255) / 256, 2048);
    cvt_transpose_bf16<<<gt, 256, 0, stream>>>(Wo, WoT, 2048, H);
  }

  dim3 g1(QKV / 128, T / 128);
  gemm_bf16_nt<<<g1, 128, 0, stream>>>(hb, WqT, qkv, T, QKV, H);

  dim3 g2(T, 24);
  normrope<<<g2, 128, 0, stream>>>(qkv, positions, qw, kw, qb, kbuf, T);
  cvt_v<<<(T * 1024 + 255) / 256, 256, 0, stream>>>(qkv, vT, T, T * 1024);

  dim3 g3(T / 64, 16);
  flash_attn<<<g3, 128, 0, stream>>>(qb, kbuf, vT, ab, T);

  dim3 g4(H / 128, T / 128);
  gemm_bf16_nt<<<g4, 128, 0, stream>>>(ab, WoT, out, T, H, 2048);
}
